// Head_23167053595343
// MI455X (gfx1250) — matrix operand order repaired and checked by reading, not compiled, not run on hardware
//
#include <hip/hip_runtime.h>

typedef __attribute__((ext_vector_type(16))) _Float16 v16h;
typedef __attribute__((ext_vector_type(4)))  _Float16 v4h;
typedef __attribute__((ext_vector_type(8)))  float    v8f;

#define B_ 8
#define T_ 2048
#define C_ 1024
#define H_ 128
#define FRAG_H 512   // halves per fragment blob (32 lanes x 32B)

#if __has_builtin(__builtin_amdgcn_global_load_async_to_lds_b128)
#define ASYNC_COPY 1
#endif

// ---- WMMA fragment loaders (contiguous k order: lane half h holds k = 16h..16h+15) ----

// A 16x32 f16, element (m,k) at base[m*stride+k]; lane M, K = 16*half + s.
// Per lane this is one contiguous 32B run -> compiler fuses to 2x b128.
__device__ inline v16h load_A_f16(const _Float16* __restrict__ base, int stride) {
  int lane = threadIdx.x & 31;
  int m = lane & 15, hh = lane >> 4;
  v16h a;
#pragma unroll
  for (int j = 0; j < 8; ++j) {
    int k = hh * 16 + 2 * j;
    a[2 * j]     = base[(size_t)m * stride + k];
    a[2 * j + 1] = base[(size_t)m * stride + k + 1];
  }
  return a;
}

// B 32x16 f16, element (k,n) at base[k*stride+n]; lane N, K split across halves.
__device__ inline v16h load_B_f16(const _Float16* __restrict__ base, int stride) {
  int lane = threadIdx.x & 31;
  int n = lane & 15, hh = lane >> 4;
  v16h b;
#pragma unroll
  for (int j = 0; j < 8; ++j) {
    int k = hh * 16 + 2 * j;
    b[2 * j]     = base[(size_t)k * stride + n];
    b[2 * j + 1] = base[(size_t)(k + 1) * stride + n];
  }
  return b;
}

// B 32x16 from transposed source: element (k,n) at base[n*stride+k].
__device__ inline v16h load_BT_f16(const _Float16* __restrict__ base, int stride) {
  int lane = threadIdx.x & 31;
  int n = lane & 15, hh = lane >> 4;
  v16h b;
#pragma unroll
  for (int j = 0; j < 8; ++j) {
    int k = hh * 16 + 2 * j;
    b[2 * j]     = base[(size_t)n * stride + k];
    b[2 * j + 1] = base[(size_t)n * stride + k + 1];
  }
  return b;
}

// B 32x16 sourced from f32 weights, converting to f16 (one-time pre-pass only).
__device__ inline v16h load_B_f32(const float* __restrict__ base, int stride) {
  int lane = threadIdx.x & 31;
  int n = lane & 15, hh = lane >> 4;
  v16h b;
#pragma unroll
  for (int j = 0; j < 8; ++j) {
    int k = hh * 16 + 2 * j;
    b[2 * j]     = (_Float16)base[(size_t)k * stride + n];
    b[2 * j + 1] = (_Float16)base[(size_t)(k + 1) * stride + n];
  }
  return b;
}

// ---- Kernel 0: one-time weight conversion f32 -> f16 B-fragment blobs ----
// wblob[mat][cchunk][htile]: mat 0=Wk,1=Wq,2=Wv; 256 frags per matrix.
__global__ void __launch_bounds__(256) conv_w(
    const float* __restrict__ Wk, const float* __restrict__ Wq,
    const float* __restrict__ Wv, _Float16* __restrict__ wblob) {
  int wv = threadIdx.x >> 5, lane = threadIdx.x & 31;
  int g = blockIdx.x * 8 + wv;           // 0..767
  int mat = g >> 8, f = g & 255;
  int cchunk = f >> 3, ht = f & 7;
  const float* W = (mat == 0) ? Wk : (mat == 1) ? Wq : Wv;
  v16h frag = load_B_f32(W + (size_t)cchunk * 32 * H_ + ht * 16, H_);
  ((v16h*)(wblob + (size_t)g * FRAG_H))[lane] = frag;
}

// ---- Kernel 1: fused Q/K/V projection, 32-row tiles, fragment-blob output ----
// Inner loop: x tile (float4 -> f16) staged once in LDS per K-step; weight
// fragments are coalesced 32B/lane blob loads; 6 WMMAs per wave per K-step.
__global__ void __launch_bounds__(256) proj_qkv(
    const float* __restrict__ x, const _Float16* __restrict__ wblob,
    _Float16* __restrict__ qblob, _Float16* __restrict__ kblob,
    _Float16* __restrict__ vblob) {
  __shared__ _Float16 xs[32 * 32];
  __shared__ _Float16 qs[32 * 128], ks[32 * 128], vs[32 * 128];
  int rt = blockIdx.x;
  int row0 = rt * 32;
  int wv = threadIdx.x >> 5;
  int h0 = wv * 16;
  int lane = threadIdx.x & 31;
  int hh = lane >> 4, nn = lane & 15;

  v8f cq[2] = {}, ck[2] = {}, cv[2] = {};
  for (int c0 = 0; c0 < C_; c0 += 32) {
    __syncthreads();
    {
      // 32x32 f32 tile: one float4 load + one packed 8B LDS store per thread
      int e = threadIdx.x * 4;
      int r = e >> 5, c = e & 31;
      const float4 xv = *(const float4*)(x + (size_t)(row0 + r) * C_ + c0 + c);
      v4h h4;
      h4[0] = (_Float16)xv.x; h4[1] = (_Float16)xv.y;
      h4[2] = (_Float16)xv.z; h4[3] = (_Float16)xv.w;
      *(v4h*)(xs + e) = h4;
    }
    __syncthreads();
    size_t fbase = ((size_t)(c0 >> 5) * 8 + wv) * FRAG_H;
    v16h bk = ((const v16h*)(wblob + 0 * 256 * FRAG_H + fbase))[lane];
    v16h bq = ((const v16h*)(wblob + 1 * 256 * FRAG_H + fbase))[lane];
    v16h bv = ((const v16h*)(wblob + 2 * 256 * FRAG_H + fbase))[lane];
#pragma unroll
    for (int mt = 0; mt < 2; ++mt) {
      v16h a = load_A_f16(xs + mt * 16 * 32, 32);
      ck[mt] = __builtin_amdgcn_wmma_f32_16x16x32_f16(false, a, false, bk, (short)0, ck[mt], false, false);
      cq[mt] = __builtin_amdgcn_wmma_f32_16x16x32_f16(false, a, false, bq, (short)0, cq[mt], false, false);
      cv[mt] = __builtin_amdgcn_wmma_f32_16x16x32_f16(false, a, false, bv, (short)0, cv[mt], false, false);
    }
  }

  // D fragments -> LDS row-major [32 t][128 h]
#pragma unroll
  for (int mt = 0; mt < 2; ++mt)
#pragma unroll
    for (int r = 0; r < 8; ++r) {
      int m = mt * 16 + r + 8 * hh;
      qs[m * 128 + h0 + nn] = (_Float16)cq[mt][r];
      ks[m * 128 + h0 + nn] = (_Float16)ck[mt][r];
      vs[m * 128 + h0 + nn] = (_Float16)cv[mt][r];
    }
  __syncthreads();

  // repack to fragment blobs (each wave emits one frag per family, coalesced 1KB)
  int b  = rt / (T_ / 32);
  int sb = rt % (T_ / 32);
  {
    int qtl = wv >> 2, kk = wv & 3;
    v16h fq = load_A_f16(qs + qtl * 16 * 128 + kk * 32, 128);
    _Float16* dst = qblob + (((size_t)b * 128 + (sb * 2 + qtl)) * 4 + kk) * FRAG_H;
    ((v16h*)dst)[lane] = fq;
  }
  {
    int stile = wv >> 2, kc = wv & 3;
    v16h fk = load_BT_f16(ks + stile * 16 * 128 + kc * 32, 128);
    _Float16* dst = kblob + (((size_t)b * 64 + sb) * 8 + (stile * 4 + kc)) * FRAG_H;
    ((v16h*)dst)[lane] = fk;
  }
  {
    v16h fv = load_B_f16(vs + wv * 16, 128);
    _Float16* dst = vblob + (((size_t)b * 64 + sb) * 8 + wv) * FRAG_H;
    ((v16h*)dst)[lane] = fv;
  }
}

// ---- cooperative K/V tile fill: 8KB K-frags + 8KB V-frags per 32-wide s-block ----
__device__ inline void fill_tile(_Float16* __restrict__ dK, const _Float16* __restrict__ sK,
                                 _Float16* __restrict__ dV, const _Float16* __restrict__ sV) {
  int t = threadIdx.x;
#ifdef ASYNC_COPY
  typedef int v4i_ __attribute__((vector_size(16)));
  typedef __attribute__((address_space(1))) v4i_ gv4i;
  typedef __attribute__((address_space(3))) v4i_ sv4i;
  gv4i* gK = (gv4i*)(unsigned long long)(sK + (size_t)t * 16);
  gv4i* gV = (gv4i*)(unsigned long long)(sV + (size_t)t * 16);
  sv4i* lK = (sv4i*)(unsigned)(unsigned long long)(dK + (size_t)t * 16);
  sv4i* lV = (sv4i*)(unsigned)(unsigned long long)(dV + (size_t)t * 16);
  __builtin_amdgcn_global_load_async_to_lds_b128(gK,     lK,     0, 0);
  __builtin_amdgcn_global_load_async_to_lds_b128(gK + 1, lK + 1, 0, 0);
  __builtin_amdgcn_global_load_async_to_lds_b128(gV,     lV,     0, 0);
  __builtin_amdgcn_global_load_async_to_lds_b128(gV + 1, lV + 1, 0, 0);
#else
  ((v16h*)dK)[t] = ((const v16h*)sK)[t];   // 256 threads x 32B = 8KB
  ((v16h*)dV)[t] = ((const v16h*)sV)[t];
#endif
}

__device__ inline void wait_fill() {
#ifdef ASYNC_COPY
#if __has_builtin(__builtin_amdgcn_s_wait_asynccnt)
  __builtin_amdgcn_s_wait_asynccnt(0);
#else
  asm volatile("s_wait_asynccnt 0x0" ::: "memory");
#endif
#endif
  __syncthreads();
}

// ---- one 32-wide key block: 8 score WMMAs + online softmax + 8 PV WMMAs ----
template <bool MASKED, class KF, class VF>
__device__ inline void process_block(const v16h (&aq)[4], KF kfrag, VF vfrag,
                                     v8f (&acc)[8], float (&mrow)[8], float (&lrow)[8],
                                     int s0, int t0, _Float16* __restrict__ pb) {
  const float scale = 0.03125f;   // C^-0.5 (reference scales by embedding size)
  int lane = threadIdx.x & 31;
  int hh = lane >> 4, nn = lane & 15;

  v8f S0 = {}, S1 = {};
#pragma unroll
  for (int kc = 0; kc < 4; ++kc) {
    v16h b0 = kfrag(kc);        // s-subtile 0
    v16h b1 = kfrag(4 + kc);    // s-subtile 1
    S0 = __builtin_amdgcn_wmma_f32_16x16x32_f16(false, aq[kc], false, b0, (short)0, S0, false, false);
    S1 = __builtin_amdgcn_wmma_f32_16x16x32_f16(false, aq[kc], false, b1, (short)0, S1, false, false);
  }
#pragma unroll
  for (int r = 0; r < 8; ++r) {
    float s0v = S0[r] * scale;
    float s1v = S1[r] * scale;
    if (MASKED) {
      int tg = t0 + r + 8 * hh;
      if (s0 + nn > tg)      s0v = -1e30f;
      if (s0 + 16 + nn > tg) s1v = -1e30f;
    }
    float rmax = fmaxf(s0v, s1v);
#pragma unroll
    for (int off = 1; off < 16; off <<= 1)
      rmax = fmaxf(rmax, __shfl_xor(rmax, off, 32));
    float mnew  = fmaxf(mrow[r], rmax);
    float alpha = __expf(mrow[r] - mnew);
    mrow[r] = mnew;
    float p0 = __expf(s0v - mnew);
    float p1 = __expf(s1v - mnew);
    float rsum = p0 + p1;
#pragma unroll
    for (int off = 1; off < 16; off <<= 1)
      rsum += __shfl_xor(rsum, off, 32);
    lrow[r] = lrow[r] * alpha + rsum;
#pragma unroll
    for (int h = 0; h < 8; ++h) acc[h][r] *= alpha;
    pb[(r + 8 * hh) * 32 + nn]      = (_Float16)p0;   // C-layout -> row-major LDS
    pb[(r + 8 * hh) * 32 + 16 + nn] = (_Float16)p1;
  }
  asm volatile("s_wait_dscnt 0x0" ::: "memory");       // cross-lane LDS dependency
  v16h pa = load_A_f16(pb, 32);
#pragma unroll
  for (int h = 0; h < 8; ++h) {
    v16h bv = vfrag(h);
    acc[h] = __builtin_amdgcn_wmma_f32_16x16x32_f16(false, pa, false, bv, (short)0, acc[h], false, false);
  }
}

// ---- Kernel 2: flash attention; block = 8 consecutive q-tiles of one batch ----
__global__ void __launch_bounds__(256) attn(
    const _Float16* __restrict__ qblob, const _Float16* __restrict__ kblob,
    const _Float16* __restrict__ vblob, float* __restrict__ out) {
  __shared__ _Float16 Kbuf[2][8 * FRAG_H];   // 16KB double-buffered K frags
  __shared__ _Float16 Vbuf[2][8 * FRAG_H];   // 16KB double-buffered V frags
  __shared__ _Float16 Pbuf[8][512];          // per-wave P transpose staging
  int wv = threadIdx.x >> 5;
  int lane = threadIdx.x & 31;
  int b  = blockIdx.x >> 4;
  int qg = blockIdx.x & 15;
  int qt = qg * 8 + wv;
  int t0 = qt * 16;

  v16h aq[4];
  {
    const v16h* qsrc = (const v16h*)(qblob + ((size_t)b * 128 + qt) * 4 * FRAG_H);
#pragma unroll
    for (int kc = 0; kc < 4; ++kc) aq[kc] = qsrc[kc * 32 + lane];
  }
  v8f acc[8];
#pragma unroll
  for (int i = 0; i < 8; ++i) acc[i] = (v8f){};
  float mrow[8], lrow[8];
#pragma unroll
  for (int r = 0; r < 8; ++r) { mrow[r] = -1e30f; lrow[r] = 0.0f; }
  _Float16* pb = Pbuf[wv];

  const _Float16* kb = kblob + (size_t)b * 64 * 8 * FRAG_H;
  const _Float16* vb = vblob + (size_t)b * 64 * 8 * FRAG_H;

  // first qg*8 key blocks: shared via LDS, unmasked
  int ncommon = qg * 8;
  if (ncommon > 0) {
    fill_tile(Kbuf[0], kb, Vbuf[0], vb);
    for (int blk = 0; blk < ncommon; ++blk) {
      int cur = blk & 1;
      wait_fill();   // own async copies done + all waves arrived
      if (blk + 1 < ncommon)
        fill_tile(Kbuf[cur ^ 1], kb + (size_t)(blk + 1) * 8 * FRAG_H,
                  Vbuf[cur ^ 1], vb + (size_t)(blk + 1) * 8 * FRAG_H);
      const _Float16* Kc = Kbuf[cur];
      const _Float16* Vc = Vbuf[cur];
      auto kf = [&](int f) { return ((const v16h*)(Kc + f * FRAG_H))[lane]; };
      auto vf = [&](int f) { return ((const v16h*)(Vc + f * FRAG_H))[lane]; };
      process_block<false>(aq, kf, vf, acc, mrow, lrow, blk * 32, t0, pb);
    }
  }

  // per-wave diagonal tail: direct coalesced blob loads + causal mask
  int nblk = (t0 + 47) >> 5;
  for (int blk = ncommon; blk < nblk; ++blk) {
    const _Float16* kc = kb + (size_t)blk * 8 * FRAG_H;
    const _Float16* vc = vb + (size_t)blk * 8 * FRAG_H;
    if (blk + 1 < nblk)
      __builtin_prefetch(kb + (size_t)(blk + 1) * 8 * FRAG_H, 0, 0);
    auto kf = [&](int f) { return ((const v16h*)(kc + f * FRAG_H))[lane]; };
    auto vf = [&](int f) { return ((const v16h*)(vc + f * FRAG_H))[lane]; };
    process_block<true>(aq, kf, vf, acc, mrow, lrow, blk * 32, t0, pb);
  }

  // epilogue: normalize, write f32 out [B][T][H]
  int hh = lane >> 4, nn = lane & 15;
#pragma unroll
  for (int r = 0; r < 8; ++r) {
    int m = r + 8 * hh;
    float inv = 1.0f / lrow[r];
    size_t row = (size_t)b * T_ + t0 + m;
#pragma unroll
    for (int h = 0; h < 8; ++h)
      out[row * H_ + h * 16 + nn] = acc[h][r] * inv;
  }
}

extern "C" void kernel_launch(void* const* d_in, const int* in_sizes, int n_in,
                              void* d_out, int out_size, void* d_ws, size_t ws_size,
                              hipStream_t stream) {
  // setup_inputs order: x, Wk, Wq, Wv (all f32)
  const float* x  = (const float*)d_in[0];
  const float* Wk = (const float*)d_in[1];
  const float* Wq = (const float*)d_in[2];
  const float* Wv = (const float*)d_in[3];
  float* out = (float*)d_out;

  _Float16* qblob = (_Float16*)d_ws;                                 // 4 MB
  _Float16* kblob = qblob + (size_t)B_ * 128 * 4 * FRAG_H;           // 4 MB
  _Float16* vblob = kblob + (size_t)B_ * 64 * 8 * FRAG_H;            // 4 MB
  _Float16* wblob = vblob + (size_t)B_ * 64 * 8 * FRAG_H;            // 0.75 MB

  conv_w<<<96, 256, 0, stream>>>(Wk, Wq, Wv, wblob);
  proj_qkv<<<B_ * T_ / 32, 256, 0, stream>>>(x, wblob, qblob, kblob, vblob);
  attn<<<B_ * 16, 256, 0, stream>>>(qblob, kblob, vblob, out);
}
